// GraphEncoder_56418690400396
// MI455X (gfx1250) — matrix operand order repaired and checked by reading, not compiled, not run on hardware
//
#include <hip/hip_runtime.h>

// Fused Catan-style graph encoder for MI455X (gfx1250, wave32, WMMA).
// One workgroup = 8 batch elements; all activations resident in LDS (f16),
// all linear layers but the fp32 readout via v_wmma_f32_16x16x32_f16 with fused bias+LN+ReLU.
// Tile features arrive via async global->LDS copy (ASYNCcnt) when available.

typedef __attribute__((ext_vector_type(8)))  _Float16 v8h;
typedef __attribute__((ext_vector_type(16))) _Float16 v16h;
typedef __attribute__((ext_vector_type(8)))  float    v8f;
typedef __attribute__((ext_vector_type(4)))  int      v4i;

#define NW  8      // waves per workgroup
#define BT  8      // batch elements per workgroup
#define LNEPS 1e-5f

// packed f16 weight-fragment offsets in d_ws (units: halves)
#define OFS_HIN 0
#define OFS_VIN 2048
#define OFS_EIN 4096
#define OFS_HU  6144     /* + rnd*8192  (K=128) */
#define OFS_VU  22528    /* + rnd*12288 (K=192) */
#define OFS_EU  47104    /* + rnd*8192  (K=128) */
#define PACKED_TOTAL 63488

#if __has_builtin(__builtin_amdgcn_global_load_async_to_lds_b128) && \
    __has_builtin(__builtin_amdgcn_s_wait_asynccnt)
#define USE_ASYNC_LDS 1
#else
#define USE_ASYNC_LDS 0
#endif

// ---------------------------------------------------------------------------
// Prep: pack fp32 weights [K,64] into B-fragment layout for wmma_f32_16x16x32_f16.
// Fragment (kc,nt) covers k in [kc*32,kc*32+32), n in [nt*16,nt*16+16):
//   lane L = ((k%16)/8)*16 + (n%16), element e = (k%8) + ((k%32)/16)*8, fragment = 512 halves.
// ---------------------------------------------------------------------------
__global__ __launch_bounds__(256) void pack_w_kernel(
    const float* __restrict__ hinW, const float* __restrict__ vinW,
    const float* __restrict__ einW, const float* __restrict__ huW,
    const float* __restrict__ vuW,  const float* __restrict__ euW,
    _Float16* __restrict__ packed)
{
  int tid = blockIdx.x * 256 + threadIdx.x;
  if (tid >= PACKED_TOTAL) return;
  const float* src; int base;
  if      (tid < 2048)  { src = hinW;        base = 0;     }
  else if (tid < 4096)  { src = vinW;        base = 2048;  }
  else if (tid < 6144)  { src = einW;        base = 4096;  }
  else if (tid < 14336) { src = huW;         base = 6144;  }
  else if (tid < 22528) { src = huW + 8192;  base = 14336; }
  else if (tid < 34816) { src = vuW;         base = 22528; }
  else if (tid < 47104) { src = vuW + 12288; base = 34816; }
  else if (tid < 55296) { src = euW;         base = 47104; }
  else                  { src = euW + 8192;  base = 55296; }
  int local = tid - base;
  int k = local >> 6, n = local & 63;
  int kc = k >> 5, kk = k & 31;
  int L  = (((kk >> 3) & 1) << 4) | (n & 15);
  int e  = (kk & 7) | ((kk >> 4) << 3);
  int nt = n >> 4;
  packed[base + (kc * 4 + nt) * 512 + L * 16 + e] = (_Float16)src[local];
}

// ---------------------------------------------------------------------------
// One wave computes a 16-row x 64-col tile:  D = relu(LN(X@W + b)*g + be).
// X: LDS f16 rows (stride xs halves), rows indexed locally from mt0*16.
// W: packed global f16 fragments. D: LDS f16, stride 64.
// A layout: lane (L&15)=row, halves {hi*8..+7} and {16+hi*8..+7} per K-chunk.
// C layout: lane holds rows r+hi*8 (r=0..7), col = nt*16 + (L&15).
// ---------------------------------------------------------------------------
__device__ __forceinline__ void layer16(
    const _Float16* __restrict__ X, int xs, int mt, int mt0, int nkc,
    const _Float16* __restrict__ W,
    const float* __restrict__ b, const float* __restrict__ g,
    const float* __restrict__ be,
    _Float16* __restrict__ D, int rowsValid, int lane)
{
  const int m = lane & 15, hi = lane >> 4;
  const _Float16* ar = X + ((mt - mt0) * 16 + m) * xs + hi * 8;
  v8f acc0 = {}, acc1 = {}, acc2 = {}, acc3 = {};
#pragma unroll
  for (int kc = 0; kc < nkc; ++kc) {
    v8h a0 = *(const v8h*)(ar + kc * 32);
    v8h a1 = *(const v8h*)(ar + kc * 32 + 16);
    v16h A = __builtin_shufflevector(a0, a1, 0,1,2,3,4,5,6,7,8,9,10,11,12,13,14,15);
    const _Float16* wb = W + kc * 2048 + lane * 16;
    v16h B0 = *(const v16h*)(wb);
    v16h B1 = *(const v16h*)(wb + 512);
    v16h B2 = *(const v16h*)(wb + 1024);
    v16h B3 = *(const v16h*)(wb + 1536);
    acc0 = __builtin_amdgcn_wmma_f32_16x16x32_f16(false, A, false, B0, (short)0, acc0, false, false);
    acc1 = __builtin_amdgcn_wmma_f32_16x16x32_f16(false, A, false, B1, (short)0, acc1, false, false);
    acc2 = __builtin_amdgcn_wmma_f32_16x16x32_f16(false, A, false, B2, (short)0, acc2, false, false);
    acc3 = __builtin_amdgcn_wmma_f32_16x16x32_f16(false, A, false, B3, (short)0, acc3, false, false);
  }
  const float b0 = b[m],  b1 = b[16 + m],  b2 = b[32 + m],  b3 = b[48 + m];
  const float g0 = g[m],  g1 = g[16 + m],  g2 = g[32 + m],  g3 = g[48 + m];
  const float e0 = be[m], e1 = be[16 + m], e2 = be[32 + m], e3 = be[48 + m];
#pragma unroll
  for (int r = 0; r < 8; ++r) {
    float y0 = acc0[r] + b0, y1 = acc1[r] + b1, y2 = acc2[r] + b2, y3 = acc3[r] + b3;
    float s  = y0 + y1 + y2 + y3;
    float s2 = y0 * y0 + y1 * y1 + y2 * y2 + y3 * y3;
#pragma unroll
    for (int msk = 1; msk <= 8; msk <<= 1) {
      s  += __shfl_xor(s,  msk, 32);
      s2 += __shfl_xor(s2, msk, 32);
    }
    float mu  = s * (1.f / 64.f);
    float var = s2 * (1.f / 64.f) - mu * mu;
    float rs  = rsqrtf(var + LNEPS);
    int orow = mt * 16 + hi * 8 + r;
    if (orow < rowsValid) {
      _Float16* dr = D + orow * 64;
      dr[m]      = (_Float16)fmaxf((y0 - mu) * rs * g0 + e0, 0.f);
      dr[16 + m] = (_Float16)fmaxf((y1 - mu) * rs * g1 + e1, 0.f);
      dr[32 + m] = (_Float16)fmaxf((y2 - mu) * rs * g2 + e2, 0.f);
      dr[48 + m] = (_Float16)fmaxf((y3 - mu) * rs * g3 + e3, 0.f);
    }
  }
}

struct GnnParams {
  const float* tf;
  const int *h2v, *v2h, *e2v, *v2e;
  const _Float16* packed;
  const float *hin_b, *hin_g, *hin_be;
  const float *vin_b, *vin_g, *vin_be;
  const float *ein_b, *ein_g, *ein_be;
  const float *hu_b, *hu_g, *hu_be;
  const float *vu_b, *vu_g, *vu_be;
  const float *eu_b, *eu_g, *eu_be;
  const float *ro_W, *ro_b, *ro_g, *ro_be;
  float* out;
};

__global__ __launch_bounds__(256) void gnn_fused(GnnParams P)
{
  // LDS: ~312 KB of the 320 KB WGP budget (1 WG / WGP)
  __shared__ __align__(16) float    s_tf32[BT * 19 * 32];    // 20 KB async staging
  __shared__ __align__(16) _Float16 s_tf  [160 * 32];        // 10 KB (152 rows + pad)
  __shared__ __align__(16) _Float16 s_hex [160 * 64];        // 20 KB
  __shared__ __align__(16) _Float16 s_vtx [2][432 * 64];     // 108 KB
  __shared__ __align__(16) _Float16 s_edge[576 * 64];        // 72 KB
  __shared__ __align__(16) _Float16 s_vraw[432 * 32];        // 27 KB
  __shared__ __align__(16) _Float16 s_X   [128 * 192];       // 48 KB (concat/staging)

  const int tid = threadIdx.x, lane = tid & 31, w = tid >> 5;
  const size_t bbase = (size_t)blockIdx.x * BT;

  // warm L2 / WGP$ with the packed B-fragment stream (global_prefetch_b8)
  if (tid < (PACKED_TOTAL * 2) / 512)
    __builtin_prefetch((const void*)((const char*)P.packed + (size_t)tid * 512), 0, 1);

  // ---- load tile features (f32 -> f16), async global->LDS when available ----
  {
    const float* src = P.tf + bbase * 19 * 32;
#if USE_ASYNC_LDS
    // 19456 B = 1216 x 16B chunks, ASYNCcnt-tracked DMA into LDS
    for (int i = tid; i < (BT * 19 * 32) / 4; i += 256) {
      __builtin_amdgcn_global_load_async_to_lds_b128(
          (v4i*)(src + i * 4), (v4i*)(s_tf32 + i * 4), 0, 0);
    }
    __builtin_amdgcn_s_wait_asynccnt(0);
    __syncthreads();
    for (int i = tid; i < BT * 19 * 32; i += 256) s_tf[i] = (_Float16)s_tf32[i];
#else
    for (int i = tid; i < BT * 19 * 32; i += 256) s_tf[i] = (_Float16)src[i];
#endif
  }
  __syncthreads();

  // ---- hex_h = LinLNReLU(tf @ hin)  rows=152, K=32 ----
  for (int mt = w; mt < 10; mt += NW)
    layer16(s_tf, 32, mt, 0, 1, P.packed + OFS_HIN, P.hin_b, P.hin_g, P.hin_be, s_hex, BT * 19, lane);
  __syncthreads();

  // ---- vertex_raw = gather(tf, v2h)  [432 x 32] ----
  for (int i = tid; i < BT * 54 * 32; i += 256) {
    int r = i >> 5, t = i & 31, v = r % 54, bi = r / 54;
    float s = 0.f; int c = 0;
#pragma unroll
    for (int k = 0; k < 3; ++k) { int a = P.v2h[v * 3 + k]; if (a >= 0) { s += (float)s_tf[(bi * 19 + a) * 32 + t]; ++c; } }
    s_vraw[i] = (_Float16)(s / (float)(c > 0 ? c : 1));
  }
  __syncthreads();

  // ---- vertex_h = LinLNReLU(vraw @ vin)  rows=432, K=32 ----
  for (int mt = w; mt < 27; mt += NW)
    layer16(s_vraw, 32, mt, 0, 1, P.packed + OFS_VIN, P.vin_b, P.vin_g, P.vin_be, s_vtx[0], BT * 54, lane);
  __syncthreads();

  // ---- edge_raw = gather(vraw, e2v) into s_X  [576 x 32] ----
  for (int i = tid; i < BT * 72 * 32; i += 256) {
    int r = i >> 5, t = i & 31, e = r % 72, bi = r / 72;
    float s = 0.f; int c = 0;
#pragma unroll
    for (int k = 0; k < 2; ++k) { int a = P.e2v[e * 2 + k]; if (a >= 0) { s += (float)s_vraw[(bi * 54 + a) * 32 + t]; ++c; } }
    s_X[i] = (_Float16)(s / (float)(c > 0 ? c : 1));
  }
  __syncthreads();

  // ---- edge_h = LinLNReLU(edge_raw @ ein)  rows=576, K=32 ----
  for (int mt = w; mt < 36; mt += NW)
    layer16(s_X, 32, mt, 0, 1, P.packed + OFS_EIN, P.ein_b, P.ein_g, P.ein_be, s_edge, BT * 72, lane);
  __syncthreads();

  // ---- message-passing rounds ----
  int cur = 0;
  for (int rnd = 0; rnd < 2; ++rnd) {
    const _Float16* pV  = s_vtx[cur];
    _Float16*       pVn = s_vtx[cur ^ 1];
    const _Float16* vuW = P.packed + OFS_VU + rnd * 12288;
    const _Float16* huW = P.packed + OFS_HU + rnd * 8192;
    const _Float16* euW = P.packed + OFS_EU + rnd * 8192;

    // vertex: [v | gather(hex) | gather(edge)] @ vu  (rows=432, K=192), chunked
    for (int mt0 = 0; mt0 < 27; mt0 += NW) {
      int rbase = mt0 * 16, rcnt = min(128, BT * 54 - rbase);
      for (int i = tid; i < rcnt * 192; i += 256) {
        int li = i / 192, c = i % 192, grow = rbase + li, v = grow % 54, bi = grow / 54;
        float val;
        if (c < 64) val = (float)pV[grow * 64 + c];
        else if (c < 128) {
          int ch = c - 64; float s = 0.f; int cn = 0;
#pragma unroll
          for (int k = 0; k < 3; ++k) { int a = P.v2h[v * 3 + k]; if (a >= 0) { s += (float)s_hex[(bi * 19 + a) * 64 + ch]; ++cn; } }
          val = s / (float)(cn > 0 ? cn : 1);
        } else {
          int ch = c - 128; float s = 0.f; int cn = 0;
#pragma unroll
          for (int k = 0; k < 3; ++k) { int a = P.v2e[v * 3 + k]; if (a >= 0) { s += (float)s_edge[(bi * 72 + a) * 64 + ch]; ++cn; } }
          val = s / (float)(cn > 0 ? cn : 1);
        }
        s_X[i] = (_Float16)val;
      }
      __syncthreads();
      int mt = mt0 + w;
      if (mt < 27)
        layer16(s_X, 192, mt, mt0, 6, vuW, P.vu_b + rnd * 64, P.vu_g + rnd * 64, P.vu_be + rnd * 64, pVn, BT * 54, lane);
      __syncthreads();
    }

    // hex: [hex | gather(oldV)] @ hu  (rows=152, K=128), in place chunk-wise
    for (int mt0 = 0; mt0 < 10; mt0 += NW) {
      int rbase = mt0 * 16, rcnt = min(128, BT * 19 - rbase);
      for (int i = tid; i < rcnt * 128; i += 256) {
        int li = i >> 7, c = i & 127, grow = rbase + li, h = grow % 19, bi = grow / 19;
        float val;
        if (c < 64) val = (float)s_hex[grow * 64 + c];
        else {
          int ch = c - 64; float s = 0.f; int cn = 0;
#pragma unroll
          for (int k = 0; k < 6; ++k) { int a = P.h2v[h * 6 + k]; if (a >= 0) { s += (float)pV[(bi * 54 + a) * 64 + ch]; ++cn; } }
          val = s / (float)(cn > 0 ? cn : 1);
        }
        s_X[i] = (_Float16)val;
      }
      __syncthreads();
      int mt = mt0 + w;
      if (mt < 10)
        layer16(s_X, 128, mt, mt0, 4, huW, P.hu_b + rnd * 64, P.hu_g + rnd * 64, P.hu_be + rnd * 64, s_hex, BT * 19, lane);
      __syncthreads();
    }

    // edge: [edge | gather(oldV)] @ eu  (rows=576, K=128), in place chunk-wise
    for (int mt0 = 0; mt0 < 36; mt0 += NW) {
      int rbase = mt0 * 16, rcnt = min(128, BT * 72 - rbase);
      for (int i = tid; i < rcnt * 128; i += 256) {
        int li = i >> 7, c = i & 127, grow = rbase + li, e = grow % 72, bi = grow / 72;
        float val;
        if (c < 64) val = (float)s_edge[grow * 64 + c];
        else {
          int ch = c - 64; float s = 0.f; int cn = 0;
#pragma unroll
          for (int k = 0; k < 2; ++k) { int a = P.e2v[e * 2 + k]; if (a >= 0) { s += (float)pV[(bi * 54 + a) * 64 + ch]; ++cn; } }
          val = s / (float)(cn > 0 ? cn : 1);
        }
        s_X[i] = (_Float16)val;
      }
      __syncthreads();
      int mt = mt0 + w;
      if (mt < 36)
        layer16(s_X, 128, mt, mt0, 4, euW, P.eu_b + rnd * 64, P.eu_g + rnd * 64, P.eu_be + rnd * 64, s_edge, BT * 72, lane);
      __syncthreads();
    }
    cur ^= 1;
  }

  // ---- readout: pooled means + 192x64 linear + LN + ReLU (fp32 VALU) ----
  const _Float16* pV = s_vtx[cur];
  float* pool = (float*)s_X;          // 8 x 192 f32
  float* yb   = pool + BT * 192;      // 8 x 64  f32
  for (int i = tid; i < BT * 192; i += 256) {
    int bi = i / 192, c = i % 192; float s = 0.f;
    if (c < 64)       { for (int h = 0; h < 19; ++h) s += (float)s_hex[(bi * 19 + h) * 64 + c];       s *= (1.f / 19.f); }
    else if (c < 128) { int ch = c - 64;  for (int v = 0; v < 54; ++v) s += (float)pV[(bi * 54 + v) * 64 + ch];   s *= (1.f / 54.f); }
    else              { int ch = c - 128; for (int e = 0; e < 72; ++e) s += (float)s_edge[(bi * 72 + e) * 64 + ch]; s *= (1.f / 72.f); }
    pool[i] = s;
  }
  __syncthreads();
  for (int i = tid; i < BT * 64; i += 256) {
    int bi = i >> 6, c = i & 63;
    float acc = P.ro_b[c];
    for (int k = 0; k < 192; ++k) acc += pool[bi * 192 + k] * P.ro_W[k * 64 + c];
    yb[i] = acc;
  }
  __syncthreads();
  {
    int bi = w;  // one wave per batch row
    float y0 = yb[bi * 64 + lane], y1 = yb[bi * 64 + 32 + lane];
    float s = y0 + y1, s2 = y0 * y0 + y1 * y1;
#pragma unroll
    for (int msk = 1; msk <= 16; msk <<= 1) { s += __shfl_xor(s, msk, 32); s2 += __shfl_xor(s2, msk, 32); }
    float mu = s * (1.f / 64.f), var = s2 * (1.f / 64.f) - mu * mu;
    float rs = rsqrtf(var + LNEPS);
    float* o = P.out + (bbase + bi) * 64;
    o[lane]      = fmaxf((y0 - mu) * rs * P.ro_g[lane]      + P.ro_be[lane],      0.f);
    o[32 + lane] = fmaxf((y1 - mu) * rs * P.ro_g[32 + lane] + P.ro_be[32 + lane], 0.f);
  }
}

extern "C" void kernel_launch(void* const* d_in, const int* in_sizes, int n_in,
                              void* d_out, int out_size, void* d_ws, size_t ws_size,
                              hipStream_t stream)
{
  const float* tf    = (const float*)d_in[0];
  const int*   h2v   = (const int*)d_in[1];
  const int*   v2h   = (const int*)d_in[2];
  const int*   e2v   = (const int*)d_in[3];
  const int*   v2e   = (const int*)d_in[4];
  const float* hinW  = (const float*)d_in[5];
  const float* vinW  = (const float*)d_in[9];
  const float* einW  = (const float*)d_in[13];
  const float* huW   = (const float*)d_in[17];
  const float* vuW   = (const float*)d_in[21];
  const float* euW   = (const float*)d_in[25];

  _Float16* packed = (_Float16*)d_ws;
  pack_w_kernel<<<(PACKED_TOTAL + 255) / 256, 256, 0, stream>>>(hinW, vinW, einW, huW, vuW, euW, packed);

  GnnParams P;
  P.tf = tf; P.h2v = h2v; P.v2h = v2h; P.e2v = e2v; P.v2e = v2e;
  P.packed = packed;
  P.hin_b = (const float*)d_in[6];  P.hin_g = (const float*)d_in[7];  P.hin_be = (const float*)d_in[8];
  P.vin_b = (const float*)d_in[10]; P.vin_g = (const float*)d_in[11]; P.vin_be = (const float*)d_in[12];
  P.ein_b = (const float*)d_in[14]; P.ein_g = (const float*)d_in[15]; P.ein_be = (const float*)d_in[16];
  P.hu_b  = (const float*)d_in[18]; P.hu_g  = (const float*)d_in[19]; P.hu_be  = (const float*)d_in[20];
  P.vu_b  = (const float*)d_in[22]; P.vu_g  = (const float*)d_in[23]; P.vu_be  = (const float*)d_in[24];
  P.eu_b  = (const float*)d_in[26]; P.eu_g  = (const float*)d_in[27]; P.eu_be  = (const float*)d_in[28];
  P.ro_W  = (const float*)d_in[29]; P.ro_b  = (const float*)d_in[30];
  P.ro_g  = (const float*)d_in[31]; P.ro_be = (const float*)d_in[32];
  P.out   = (float*)d_out;

  int B = in_sizes[0] / (19 * 32);       // 16384
  gnn_fused<<<B / BT, 256, 0, stream>>>(P);
}
